// EncoderDecoder_21706764714370
// MI455X (gfx1250) — matrix operand order repaired and checked by reading, not compiled, not run on hardware
//
#include <hip/hip_runtime.h>
#include <math.h>

// ---------------- constants (fixed shapes from the reference) ----------------
static constexpr int Bc   = 32;
static constexpr int Nc   = 16384;
static constexpr int Tc   = 12;
static constexpr int Ec   = 65536;
static constexpr int Rc   = 200;
static constexpr int REc  = 800;
static constexpr int INc  = 64;
static constexpr int Hc   = 128;     // OSD = OTD = SCD = 128
static constexpr int TCDc = 64;
static constexpr int SFDc = 32;
static constexpr int WKc  = 6;
static constexpr int DATT = 256;
static constexpr long NT  = (long)Nc * Tc;

typedef __attribute__((ext_vector_type(16))) __bf16 v16bf;
typedef __attribute__((ext_vector_type(8)))  __bf16 v8bf;
typedef __attribute__((ext_vector_type(8)))  float  v8f;

__device__ __forceinline__ float gelu_f(float x) {
    return 0.5f * x * (1.0f + erff(x * 0.70710678118654752f));
}
__device__ __forceinline__ float sigm_f(float x) {
    return 1.0f / (1.0f + expf(-x));
}

// ---------------- elementwise / utility kernels ----------------
__global__ void k_zero(float* p, long n) {
    long i = (long)blockIdx.x * blockDim.x + threadIdx.x;
    if (i < n) p[i] = 0.0f;
}
__global__ void k_zero_bf(__bf16* p, long n) {
    long i = (long)blockIdx.x * blockDim.x + threadIdx.x;
    if (i < n) p[i] = (__bf16)0.0f;
}
__global__ void k_cast_bf(const float* __restrict__ x, __bf16* __restrict__ y, long n) {
    long i = (long)blockIdx.x * blockDim.x + threadIdx.x;
    if (i < n) y[i] = (__bf16)x[i];
}
__global__ void k_scale(const float* __restrict__ x, float s, float* __restrict__ y, long n) {
    long i = (long)blockIdx.x * blockDim.x + threadIdx.x;
    if (i < n) y[i] = x[i] * s;
}
__global__ void k_add(const float* __restrict__ a, const float* __restrict__ b,
                      float* __restrict__ y, long n) {
    long i = (long)blockIdx.x * blockDim.x + threadIdx.x;
    if (i < n) y[i] = a[i] + b[i];
}

// ---------------- weight repack to WMMA B-fragment layout ----------------
// packed[((kt*ntiles + nt)*32 + lane)*16 + e] maps to
// k_local = (e&7) + 8*(lane>>4) + 16*(e>>3), n = nt*16 + (lane&15)
__global__ void k_repack(const float* __restrict__ W, __bf16* __restrict__ P,
                         int K, int Nw, int trans) {
    long idx = (long)blockIdx.x * blockDim.x + threadIdx.x;
    long total = (long)K * Nw;
    if (idx >= total) return;
    int e    = (int)(idx & 15);
    int lane = (int)((idx >> 4) & 31);
    long r   = idx >> 9;
    int ntiles = Nw >> 4;
    int nt = (int)(r % ntiles);
    int kt = (int)(r / ntiles);
    int j = e >> 1, p = e & 1;
    int k = kt * 32 + (e & 7) + 8 * (lane >> 4) + 16 * (e >> 3);
    int n = nt * 16 + (lane & 15);
    float v = trans ? W[(long)n * K + k] : W[(long)k * Nw + n];
    P[idx] = (__bf16)v;
}

// ---- bf16 WMMA GEMM: C = act(A @ B + bias*mult); 32x16 output per wave ----
// A-fragment per the CDNA5 16-bit A layout: per lane two contiguous 8-elem
// runs at kt*32 + 8*half and kt*32 + 16 + 8*half -> two b128 loads.
// Out-of-range rows read from a zeroed dummy row (no per-element masking).
__global__ __launch_bounds__(256)
void k_wmma_gemm(const __bf16* __restrict__ A, long lda,
                 const __bf16* __restrict__ P,
                 const __bf16* __restrict__ zrow,
                 const float* __restrict__ bias, float bias_mult,
                 float* __restrict__ C, __bf16* __restrict__ Cbf, long ldc,
                 int M, int K, int Nw, int act) {
    int wave = threadIdx.x >> 5;
    int lane = threadIdx.x & 31;
    int ntiles = Nw >> 4;
    int nt = blockIdx.y * 8 + wave;
    if (nt >= ntiles) return;               // wave-uniform exit
    int m0   = blockIdx.x << 5;             // 2 m-tiles per wave
    int half = lane >> 4;
    int mr   = lane & 15;
    int mA = m0 + mr;
    int mB = m0 + 16 + mr;
    const __bf16* Arow0 = (mA < M) ? (A + (long)mA * lda) : zrow;
    const __bf16* Arow1 = (mB < M) ? (A + (long)mB * lda) : zrow;
    v8f acc0 = {};
    v8f acc1 = {};
    int ktiles = K >> 5;
    const __bf16* Pbase = P + ((long)nt * 32 + lane) * 16;
    long pstride = (long)ntiles * 512;      // 32 lanes * 16 elems
    int aoff = 8 * half;
    for (int kt = 0; kt < ktiles; ++kt) {
        int kb = kt * 32 + aoff;
        v8bf lo0 = *(const v8bf*)(Arow0 + kb);
        v8bf hi0 = *(const v8bf*)(Arow0 + kb + 16);
        v8bf lo1 = *(const v8bf*)(Arow1 + kb);
        v8bf hi1 = *(const v8bf*)(Arow1 + kb + 16);
        v16bf a0 = __builtin_shufflevector(lo0, hi0, 0,1,2,3,4,5,6,7,8,9,10,11,12,13,14,15);
        v16bf a1 = __builtin_shufflevector(lo1, hi1, 0,1,2,3,4,5,6,7,8,9,10,11,12,13,14,15);
        v16bf b = *(const v16bf*)(Pbase + (long)kt * pstride);
        acc0 = __builtin_amdgcn_wmma_f32_16x16x32_bf16(
                   false, a0, false, b, (short)0, acc0, false, false);
        acc1 = __builtin_amdgcn_wmma_f32_16x16x32_bf16(
                   false, a1, false, b, (short)0, acc1, false, false);
    }
    int n = nt * 16 + mr;
    float bv = bias ? bias[n] * bias_mult : 0.0f;
#pragma unroll
    for (int r = 0; r < 8; ++r) {
        int mm = m0 + r + 8 * half;
        if (mm < M) {
            float v = acc0[r] + bv;
            if (act == 1) v = gelu_f(v);
            if (C)   C[(long)mm * ldc + n] = v;
            if (Cbf) Cbf[(long)mm * ldc + n] = (__bf16)v;
        }
    }
#pragma unroll
    for (int r = 0; r < 8; ++r) {
        int mm = m0 + 16 + r + 8 * half;
        if (mm < M) {
            float v = acc1[r] + bv;
            if (act == 1) v = gelu_f(v);
            if (C)   C[(long)mm * ldc + n] = v;
            if (Cbf) Cbf[(long)mm * ldc + n] = (__bf16)v;
        }
    }
}

// ---------------- naive fp32 GEMM for tiny matrices ----------------
__global__ void k_naive_gemm(const float* __restrict__ A, long lda,
                             const float* __restrict__ Bm, int btrans,
                             const float* __restrict__ bias,
                             float* __restrict__ C, long ldc,
                             int M, int K, int Nw, int act) {
    long idx = (long)blockIdx.x * blockDim.x + threadIdx.x;
    if (idx >= (long)M * Nw) return;
    int m = (int)(idx / Nw), n = (int)(idx % Nw);
    float s = bias ? bias[n] : 0.0f;
    const float* ar = A + (long)m * lda;
    if (btrans) {
        const float* br = Bm + (long)n * K;
        for (int k = 0; k < K; ++k) s += ar[k] * br[k];
    } else {
        for (int k = 0; k < K; ++k) s += ar[k] * Bm[(long)k * Nw + n];
    }
    if (act == 1) s = gelu_f(s);
    C[(long)m * ldc + n] = s;
}

// ---------------- graph-conv pieces ----------------
__global__ void k_deg(const int* __restrict__ src, const int* __restrict__ dst,
                      float* __restrict__ dego, float* __restrict__ degi, int E) {
    int e = blockIdx.x * blockDim.x + threadIdx.x;
    if (e < E) {
        atomicAdd(&dego[src[e]], 1.0f);
        atomicAdd(&degi[dst[e]], 1.0f);
    }
}
__global__ void k_scalecast(const float* __restrict__ x, const float* __restrict__ deg,
                            __bf16* __restrict__ y, int F, long total) {
    long i = (long)blockIdx.x * blockDim.x + threadIdx.x;
    if (i >= total) return;
    long n = i / F;
    float d = deg[n]; if (d < 1.0f) d = 1.0f;
    y[i] = (__bf16)(x[i] * rsqrtf(d));
}
__global__ void k_scalecast_lsh(const float* __restrict__ lsf,
                                const float* __restrict__ link_tab,
                                const int* __restrict__ lsi,
                                const float* __restrict__ deg,
                                __bf16* __restrict__ y) {
    long i = (long)blockIdx.x * blockDim.x + threadIdx.x;
    if (i >= (long)Nc * 160) return;
    int n = (int)(i / 160), f = (int)(i % 160);
    float v = (f < SFDc) ? lsf[(long)n * SFDc + f]
                         : link_tab[(long)lsi[n] * Hc + (f - SFDc)];
    float d = deg[n]; if (d < 1.0f) d = 1.0f;
    y[i] = (__bf16)(v * rsqrtf(d));
}
__global__ void k_scalecast_htd(const float* __restrict__ ht,
                                const float* __restrict__ chs,
                                const float* __restrict__ deg,
                                __bf16* __restrict__ y) {
    long i = (long)blockIdx.x * blockDim.x + threadIdx.x;
    if (i >= (long)Nc * 256) return;
    int n = (int)(i / 256), f = (int)(i % 256);
    float v = (f < Hc) ? ht[(long)n * Hc + f] : chs[(long)n * Hc + (f - Hc)];
    float d = deg[n]; if (d < 1.0f) d = 1.0f;
    y[i] = (__bf16)(v * rsqrtf(d));
}
__global__ void k_scatter(const int* __restrict__ src, const int* __restrict__ dst,
                          const float* __restrict__ h, float* __restrict__ agg, int F) {
    int e = blockIdx.x;
    long sb = (long)src[e] * F, db = (long)dst[e] * F;
    for (int i = threadIdx.x; i < F; i += blockDim.x)
        atomicAdd(&agg[db + i], h[sb + i]);
}
__global__ void k_conv_final(const float* __restrict__ agg, const float* __restrict__ degi,
                             const float* __restrict__ b, float* __restrict__ out,
                             int Dout, int F, long total, int act) {
    long i = (long)blockIdx.x * blockDim.x + threadIdx.x;
    if (i >= total) return;
    long n = i / F;
    int d = (int)(i % Dout);
    float dg = degi[n]; if (dg < 1.0f) dg = 1.0f;
    float v = agg[i] * rsqrtf(dg) + b[d];
    if (act == 1) v = gelu_f(v);
    out[i] = v;
}

// ---------------- segment ops ----------------
__global__ void k_seg_counts(const int* __restrict__ lbi, float* __restrict__ cnt, int N) {
    int n = blockIdx.x * blockDim.x + threadIdx.x;
    if (n < N) atomicAdd(&cnt[lbi[n]], 1.0f);
}
__global__ void k_seg_sum(const float* __restrict__ x, const int* __restrict__ lbi,
                          float* __restrict__ out, int F) {
    int n = blockIdx.x;
    long ob = (long)lbi[n] * F, xb = (long)n * F;
    for (int i = threadIdx.x; i < F; i += blockDim.x)
        atomicAdd(&out[ob + i], x[xb + i]);
}
__global__ void k_seg_div(float* __restrict__ hs, const float* __restrict__ cnt,
                          int F, int Bn) {
    long i = (long)blockIdx.x * blockDim.x + threadIdx.x;
    if (i >= (long)Bn * F) return;
    hs[i] /= cnt[i / F];
}
__global__ void k_mean_T(const float* __restrict__ x, float* __restrict__ out,
                         int rows, int D) {
    long i = (long)blockIdx.x * blockDim.x + threadIdx.x;
    if (i >= (long)rows * D) return;
    int r = (int)(i / D), d = (int)(i % D);
    float s = 0.0f;
    for (int t = 0; t < Tc; ++t) s += x[((long)r * Tc + t) * D + d];
    out[i] = s * (1.0f / Tc);
}

// ---------------- LSTM pieces ----------------
__global__ void k_lstm_cell(const float* __restrict__ g1, const float* __restrict__ g2,
                            const float* __restrict__ bih, const float* __restrict__ bhh,
                            float* __restrict__ h, float* __restrict__ c,
                            __bf16* __restrict__ hbf, float* __restrict__ hsum,
                            long Nrow) {
    long idx = (long)blockIdx.x * blockDim.x + threadIdx.x;
    if (idx >= Nrow * Hc) return;
    long n = idx / Hc; int j = (int)(idx % Hc);
    long base = n * 4 * Hc;
    float gi = g1[base + j]          + g2[base + j]          + bih[j]          + bhh[j];
    float gf = g1[base + Hc + j]     + g2[base + Hc + j]     + bih[Hc + j]     + bhh[Hc + j];
    float gg = g1[base + 2*Hc + j]   + g2[base + 2*Hc + j]   + bih[2*Hc + j]   + bhh[2*Hc + j];
    float go = g1[base + 3*Hc + j]   + g2[base + 3*Hc + j]   + bih[3*Hc + j]   + bhh[3*Hc + j];
    float cc = sigm_f(gf) * c[idx] + sigm_f(gi) * tanhf(gg);
    float hh = sigm_f(go) * tanhf(cc);
    c[idx] = cc; h[idx] = hh; hbf[idx] = (__bf16)hh; hsum[idx] += hh;
}

// one block per batch row; blockDim = 512 (= 4*H); whole sequence in one launch
__global__ __launch_bounds__(512)
void k_small_lstm(const float* __restrict__ x, int In,
                  const float* __restrict__ Wih, const float* __restrict__ Whh,
                  const float* __restrict__ bih, const float* __restrict__ bhh,
                  float* __restrict__ hs_out) {
    __shared__ float g[512];
    __shared__ float hbuf[Hc];
    __shared__ float cbuf[Hc];
    __shared__ float xbuf[192];
    int b = blockIdx.x;
    int j = threadIdx.x;
    if (j < Hc) { hbuf[j] = 0.0f; cbuf[j] = 0.0f; }
    __syncthreads();
    for (int t = 0; t < Tc; ++t) {
        for (int i = j; i < In; i += 512) xbuf[i] = x[((long)b * Tc + t) * In + i];
        __syncthreads();
        float s = bih[j] + bhh[j];
        const float* wi = Wih + (long)j * In;
        for (int k = 0; k < In; ++k) s += xbuf[k] * wi[k];
        const float* wh = Whh + (long)j * Hc;
        for (int k = 0; k < Hc; ++k) s += hbuf[k] * wh[k];
        g[j] = s;
        __syncthreads();
        if (j < Hc) {
            float ig = sigm_f(g[j]);
            float fg = sigm_f(g[Hc + j]);
            float gg = tanhf(g[2 * Hc + j]);
            float og = sigm_f(g[3 * Hc + j]);
            float c = fg * cbuf[j] + ig * gg;
            float h = og * tanhf(c);
            cbuf[j] = c; hbuf[j] = h;
            hs_out[((long)b * Tc + t) * Hc + j] = h;
        }
        __syncthreads();
    }
}

// ---------------- gathers / concats / attention ----------------
__global__ void k_emb_gather(const int* __restrict__ weeks, const int* __restrict__ minutes,
                             const float* __restrict__ wtab, const float* __restrict__ mtab,
                             float* __restrict__ emb) {
    long i = (long)blockIdx.x * blockDim.x + threadIdx.x;
    if (i >= (long)Bc * Tc * 70) return;
    int bt = (int)(i / 70), f = (int)(i % 70);
    emb[i] = (f < WKc) ? wtab[weeks[bt] * WKc + f]
                       : mtab[(long)minutes[bt] * TCDc + (f - WKc)];
}
__global__ void k_gather_rows(const float* __restrict__ tab, const int* __restrict__ idx,
                              float* __restrict__ out, int D, int rows) {
    long i = (long)blockIdx.x * blockDim.x + threadIdx.x;
    if (i >= (long)rows * D) return;
    int r = (int)(i / D), d = (int)(i % D);
    out[i] = tab[(long)idx[r] * D + d];
}
__global__ void k_cht_add(const float* __restrict__ eto, const float* __restrict__ eso,
                          float* __restrict__ cht) {
    long i = (long)blockIdx.x * blockDim.x + threadIdx.x;
    if (i >= (long)Bc * Tc * TCDc) return;
    int bt = (int)(i / TCDc), d = (int)(i % TCDc);
    int b = bt / Tc, t = bt % Tc;
    cht[i] = eto[i] + eso[(long)b * (TCDc * Tc) + t * TCDc + d];
}
__global__ void k_chs(const float* __restrict__ gth, const int* __restrict__ lbi,
                      const float* __restrict__ lsh, float* __restrict__ chs) {
    long i = (long)blockIdx.x * blockDim.x + threadIdx.x;
    if (i >= (long)Nc * Hc) return;
    int n = (int)(i / Hc), d = (int)(i % Hc);
    chs[i] = gth[(long)lbi[n] * Hc + d] + lsh[i];
}
__global__ void k_concat_hsd(const float* __restrict__ hs, const float* __restrict__ cht,
                             float* __restrict__ hsd) {
    long i = (long)blockIdx.x * blockDim.x + threadIdx.x;
    if (i >= (long)Bc * Tc * 192) return;
    int bt = (int)(i / 192), f = (int)(i % 192);
    hsd[i] = (f < Hc) ? hs[(long)bt * Hc + f] : cht[(long)bt * TCDc + (f - Hc)];
}
__global__ void k_beta(const float* __restrict__ htd2, const float* __restrict__ ht,
                       float* __restrict__ beta, __bf16* __restrict__ bbf) {
    long i = (long)blockIdx.x * blockDim.x + threadIdx.x;
    if (i >= (long)Nc * Hc) return;
    float v = htd2[i] + ht[i];
    beta[i] = v; bbf[i] = (__bf16)v;
}
__global__ void k_asum_bf(const float* __restrict__ alpha, const int* __restrict__ lbi,
                          __bf16* __restrict__ asum) {
    long i = (long)blockIdx.x * blockDim.x + threadIdx.x;
    if (i >= (long)Nc * Hc) return;
    int n = (int)(i / Hc), k = (int)(i % Hc);
    long base = (long)lbi[n] * Tc * Hc + k;
    float s = 0.0f;
    for (int t = 0; t < Tc; ++t) s += alpha[base + (long)t * Hc];
    asum[i] = (__bf16)s;
}
__global__ void k_scores(const float* __restrict__ alpha, const int* __restrict__ lbi,
                         const float* __restrict__ u, const float* __restrict__ ka,
                         const float* __restrict__ bq, float* __restrict__ w) {
    __shared__ float s[Tc];
    __shared__ float bqk;
    int n = blockIdx.x;
    int t = threadIdx.x;
    if (t == 0) {
        float acc = 0.0f;
        const float* kr = ka + (long)n * DATT;
        for (int d = 0; d < DATT; ++d) acc += bq[d] * kr[d];
        bqk = acc;
    }
    if (t < Tc) {
        const float* ar = alpha + ((long)lbi[n] * Tc + t) * Hc;
        const float* ur = u + (long)n * Hc;
        float acc = 0.0f;
        for (int k = 0; k < Hc; ++k) acc += ar[k] * ur[k];
        s[t] = acc;
    }
    __syncthreads();
    if (t == 0) {
        float mx = s[0] + bqk;
        for (int i = 1; i < Tc; ++i) { float v = s[i] + bqk; if (v > mx) mx = v; }
        float e[Tc]; float sum = 0.0f;
        for (int i = 0; i < Tc; ++i) { e[i] = expf(s[i] + bqk - mx); sum += e[i]; }
        for (int i = 0; i < Tc; ++i) w[(long)n * Tc + i] = e[i] / sum;
    }
}
__global__ void k_final_out(float* __restrict__ out, const float* __restrict__ w,
                            const float* __restrict__ vo, const float* __restrict__ bo) {
    int nt = blockIdx.x;                // n*Tc + t
    int n = nt / Tc;
    int d = threadIdx.x;                // 0..255
    out[(long)nt * DATT + d] += w[nt] * vo[(long)n * DATT + d] + bo[d];
}

// ---------------- host side ----------------
static inline unsigned gdiv(long n, long b) { return (unsigned)((n + b - 1) / b); }

extern "C" void kernel_launch(void* const* d_in, const int* in_sizes, int n_in,
                              void* d_out, int out_size, void* d_ws, size_t ws_size,
                              hipStream_t stream) {
    (void)in_sizes; (void)n_in; (void)out_size;
    // inputs
    const int*   weeks   = (const int*)d_in[0];
    const int*   minutes = (const int*)d_in[1];
    const int*   gsi     = (const int*)d_in[2];
    const int*   src     = (const int*)d_in[3];
    const int*   dst     = (const int*)d_in[4];
    const int*   lbi     = (const int*)d_in[5];
    const int*   lsi     = (const int*)d_in[6];
    const int*   rsrc    = (const int*)d_in[7];
    const int*   rdst    = (const int*)d_in[8];
    const float* traffic = (const float*)d_in[9];
    const float* lsf     = (const float*)d_in[10];
    const float* enc_W0  = (const float*)d_in[11];
    const float* enc_b0  = (const float*)d_in[12];
    const float* enc_W1  = (const float*)d_in[13];
    const float* enc_b1  = (const float*)d_in[14];
    const float* enc_Wih = (const float*)d_in[15];
    const float* enc_Whh = (const float*)d_in[16];
    const float* enc_bih = (const float*)d_in[17];
    const float* enc_bhh = (const float*)d_in[18];
    const float* week_tab  = (const float*)d_in[19];
    const float* minute_tab= (const float*)d_in[20];
    const float* tr_Wih  = (const float*)d_in[21];
    const float* tr_Whh  = (const float*)d_in[22];
    const float* tr_bih  = (const float*)d_in[23];
    const float* tr_bhh  = (const float*)d_in[24];
    const float* region_tab = (const float*)d_in[25];
    const float* gg_W    = (const float*)d_in[26];
    const float* gg_b    = (const float*)d_in[27];
    const float* link_tab= (const float*)d_in[28];
    const float* lg_W    = (const float*)d_in[29];
    const float* lg_b    = (const float*)d_in[30];
    const float* et_W1   = (const float*)d_in[31];
    const float* et_b1   = (const float*)d_in[32];
    const float* et_W2   = (const float*)d_in[33];
    const float* et_b2   = (const float*)d_in[34];
    const float* es_W1   = (const float*)d_in[35];
    const float* es_b1   = (const float*)d_in[36];
    const float* es_W2   = (const float*)d_in[37];
    const float* es_b2   = (const float*)d_in[38];
    const float* dec_W0  = (const float*)d_in[39];
    const float* dec_b0  = (const float*)d_in[40];
    const float* dec_W1  = (const float*)d_in[41];
    const float* dec_b1  = (const float*)d_in[42];
    const float* dec_Wih = (const float*)d_in[43];
    const float* dec_Whh = (const float*)d_in[44];
    const float* dec_bih = (const float*)d_in[45];
    const float* dec_bhh = (const float*)d_in[46];
    const float* eb_W1   = (const float*)d_in[47];
    const float* eb_b1   = (const float*)d_in[48];
    const float* eb_W2   = (const float*)d_in[49];
    const float* eb_b2   = (const float*)d_in[50];
    const float* s2t_Wv  = (const float*)d_in[55];
    const float* s2t_bv  = (const float*)d_in[56];
    const float* s2t_Wo  = (const float*)d_in[57];
    const float* s2t_bo  = (const float*)d_in[58];
    const float* t2s_Wq  = (const float*)d_in[59];
    const float* t2s_bq  = (const float*)d_in[60];
    const float* t2s_Wk  = (const float*)d_in[61];
    const float* t2s_bk  = (const float*)d_in[62];
    const float* t2s_Wv  = (const float*)d_in[63];
    const float* t2s_bv  = (const float*)d_in[64];
    const float* t2s_Wo  = (const float*)d_in[65];
    const float* t2s_bo  = (const float*)d_in[66];
    float* out = (float*)d_out;

    // -------- deterministic bump allocator over d_ws --------
    size_t off = 0;
    auto alloc = [&](size_t bytes) -> char* {
        char* p = (char*)d_ws + off;
        off += (bytes + 255) & ~(size_t)255;
        return p;
    };
    // packed weights
    struct PK { const float* W; int K, Nw, tr; __bf16* P; };
    PK pk[16] = {
        {enc_W0, 64, 128, 0, nullptr},  {enc_W1, 128, 128, 0, nullptr},
        {enc_Wih, 64, 512, 1, nullptr}, {enc_Whh, 128, 512, 1, nullptr},
        {gg_W, 128, 128, 0, nullptr},   {lg_W, 160, 128, 0, nullptr},
        {dec_W0, 256, 128, 0, nullptr}, {dec_W1, 128, 128, 0, nullptr},
        {s2t_Wv, 128, 256, 0, nullptr}, {s2t_Wo, 256, 256, 0, nullptr},
        {t2s_Wk, 128, 256, 0, nullptr}, {t2s_Wv, 128, 256, 0, nullptr},
        {t2s_Wq, 256, 128, 1, nullptr}, {t2s_Wo, 256, 256, 0, nullptr},
        {eb_W1, 256, 256, 0, nullptr},  {eb_W2, 256, 3072, 0, nullptr},
    };
    for (int i = 0; i < 16; ++i)
        pk[i].P = (__bf16*)alloc((size_t)pk[i].K * pk[i].Nw * sizeof(__bf16));
    __bf16* zrow  = (__bf16*)alloc(256 * sizeof(__bf16));
    float* deg_out = (float*)alloc(Nc * sizeof(float));
    float* deg_in  = (float*)alloc(Nc * sizeof(float));
    float* rdego   = (float*)alloc(Rc * sizeof(float));
    float* rdegi   = (float*)alloc(Rc * sizeof(float));
    float* counts  = (float*)alloc(Bc * sizeof(float));
    const long CH = (long)Nc * Hc;                 // N*128 chunk
    float* BIGh = (float*)alloc(NT * 128 * sizeof(float));
    float* BIGa = (float*)alloc(NT * 128 * sizeof(float));
    float* BIGx = (float*)alloc(NT * 128 * sizeof(float));
    __bf16* XBF = (__bf16*)alloc(NT * 192 * sizeof(__bf16));   // bf16 pool
    __bf16* TBF = (__bf16*)alloc(NT * 64 * sizeof(__bf16));    // traffic bf16
    float* G1   = (float*)alloc((long)Nc * 512 * sizeof(float));
    float* G2   = (float*)alloc((long)Nc * 512 * sizeof(float));
    float* hcur = (float*)alloc(CH * sizeof(float));
    float* ccur = (float*)alloc(CH * sizeof(float));
    float* hsum = (float*)alloc(CH * sizeof(float));
    float* ht   = (float*)alloc(CH * sizeof(float));
    __bf16* hbf = (__bf16*)alloc(CH * sizeof(__bf16));
    // small buffers
    float* emb  = (float*)alloc((long)Bc * Tc * 70 * sizeof(float));
    float* lth  = (float*)alloc((long)Bc * Tc * Hc * sizeof(float));
    float* gth  = (float*)alloc((long)Bc * Hc * sizeof(float));
    __bf16* rxs = (__bf16*)alloc((long)Rc * Hc * sizeof(__bf16));
    float* rh   = (float*)alloc((long)Rc * Hc * sizeof(float));
    float* ragg = (float*)alloc((long)Rc * Hc * sizeof(float));
    float* gemb = (float*)alloc((long)Rc * Hc * sizeof(float));
    float* gsh  = (float*)alloc((long)Bc * Hc * sizeof(float));
    float* eth  = (float*)alloc((long)Bc * Tc * 256 * sizeof(float));
    float* eto  = (float*)alloc((long)Bc * Tc * TCDc * sizeof(float));
    float* esh  = (float*)alloc((long)Bc * 256 * sizeof(float));
    float* eso  = (float*)alloc((long)Bc * 768 * sizeof(float));
    float* cht  = (float*)alloc((long)Bc * Tc * TCDc * sizeof(float));
    float* hsd  = (float*)alloc((long)Bc * Tc * 192 * sizeof(float));
    float* decH = (float*)alloc((long)Bc * Tc * Hc * sizeof(float));
    float* alpha= (float*)alloc((long)Bc * Tc * Hc * sizeof(float));
    float* hsB  = (float*)alloc((long)Bc * Tc * Hc * sizeof(float));
    if (off > ws_size) return;   // insufficient workspace; bail deterministically

    // aliases into big pools (stage ordering keeps these disjoint in time)
    float* lshb = BIGx + 0 * CH;
    float* c_hs = BIGx + 1 * CH;
    float* htd1 = BIGx + 2 * CH;
    float* htd2 = BIGx + 3 * CH;
    float* beta = BIGx + 4 * CH;
    float* ka   = BIGh + 0 * CH;       // N x 256
    float* va   = BIGh + 2 * CH;       // N x 256
    float* ubuf = BIGh + 4 * CH;       // N x 128
    float* vo   = BIGh + 5 * CH;       // N x 256
    float* vbs  = BIGh + 7 * CH;       // N x 256
    float* wbuf = BIGh + 9 * CH;       // N x T
    __bf16* xb = XBF;
    __bf16* xb_lsh  = xb; xb += (long)Nc * 160;
    __bf16* xb_htd  = xb; xb += (long)Nc * 256;
    __bf16* xb_dc   = xb; xb += (long)Nc * 128;
    __bf16* xb_beta = xb; xb += (long)Nc * 128;
    __bf16* ka_bf   = xb; xb += (long)Nc * 256;
    __bf16* va_bf   = xb; xb += (long)Nc * 256;
    __bf16* asum_bf = xb; xb += (long)Nc * 128;
    __bf16* vbs_bf  = xb; xb += (long)Nc * 256;
    __bf16* bnn_bf  = xb; xb += (long)Nc * 256;
    __bf16* hb_bf   = xb; xb += (long)Nc * 256;

    auto zero = [&](float* p, long n) { k_zero<<<gdiv(n,256),256,0,stream>>>(p, n); };
    auto gemm = [&](const __bf16* A, long lda, const __bf16* P, const float* bias,
                    float bmult, float* C, __bf16* Cbf, long ldc,
                    int M, int K, int Nw, int act) {
        dim3 g(gdiv(M,32), gdiv(Nw/16, 8));
        k_wmma_gemm<<<g, 256, 0, stream>>>(A, lda, P, zrow, bias, bmult, C, Cbf, ldc, M, K, Nw, act);
    };

    // -------- 0) repack weights; zero dummy row --------
    for (int i = 0; i < 16; ++i) {
        long tot = (long)pk[i].K * pk[i].Nw;
        k_repack<<<gdiv(tot,256),256,0,stream>>>(pk[i].W, pk[i].P, pk[i].K, pk[i].Nw, pk[i].tr);
    }
    k_zero_bf<<<1,256,0,stream>>>(zrow, 256);
    __bf16 *P_encW0=pk[0].P,*P_encW1=pk[1].P,*P_WihT=pk[2].P,*P_WhhT=pk[3].P,
           *P_ggW=pk[4].P,*P_lgW=pk[5].P,*P_decW0=pk[6].P,*P_decW1=pk[7].P,
           *P_s2tWv=pk[8].P,*P_s2tWo=pk[9].P,*P_t2sWk=pk[10].P,*P_t2sWv=pk[11].P,
           *P_t2sWqT=pk[12].P,*P_t2sWo=pk[13].P,*P_ebW1=pk[14].P,*P_ebW2=pk[15].P;

    // -------- 1) degrees for main graph --------
    zero(deg_out, Nc); zero(deg_in, Nc);
    k_deg<<<gdiv(Ec,256),256,0,stream>>>(src, dst, deg_out, deg_in, Ec);

    // -------- 2) encoder graph conv 0 (N,T,64) -> (N,T,128) --------
    k_scalecast<<<gdiv(NT*64,256),256,0,stream>>>(traffic, deg_out, XBF, Tc*INc, NT*64);
    gemm(XBF, 64, P_encW0, nullptr, 1.f, BIGh, nullptr, 128, (int)NT, 64, 128, 0);
    zero(BIGa, NT * 128);
    k_scatter<<<Ec,256,0,stream>>>(src, dst, BIGh, BIGa, Tc*128);
    k_conv_final<<<gdiv(NT*128,256),256,0,stream>>>(BIGa, deg_in, enc_b0, BIGx, 128, Tc*128, NT*128, 1);

    // -------- 3) encoder graph conv 1 --------
    k_scalecast<<<gdiv(NT*128,256),256,0,stream>>>(BIGx, deg_out, XBF, Tc*128, NT*128);
    gemm(XBF, 128, P_encW1, nullptr, 1.f, BIGh, nullptr, 128, (int)NT, 128, 128, 0);
    zero(BIGa, NT * 128);
    k_scatter<<<Ec,256,0,stream>>>(src, dst, BIGh, BIGa, Tc*128);
    k_conv_final<<<gdiv(NT*128,256),256,0,stream>>>(BIGa, deg_in, enc_b1, BIGx, 128, Tc*128, NT*128, 1);

    // -------- 4) hs = segment mean --------
    zero(counts, Bc);
    k_seg_counts<<<gdiv(Nc,256),256,0,stream>>>(lbi, counts, Nc);
    zero(hsB, (long)Bc * Tc * Hc);
    k_seg_sum<<<Nc,256,0,stream>>>(BIGx, lbi, hsB, Tc*Hc);
    k_seg_div<<<gdiv((long)Bc*Tc*Hc,256),256,0,stream>>>(hsB, counts, Tc*Hc, Bc);

    // -------- 5) encoder LSTM over raw traffic; ht = mean_t(h) --------
    k_cast_bf<<<gdiv(NT*64,256),256,0,stream>>>(traffic, TBF, NT*64);
    zero(ccur, CH); zero(hsum, CH);
    k_zero_bf<<<gdiv(CH,256),256,0,stream>>>(hbf, CH);
    for (int t = 0; t < Tc; ++t) {
        gemm(TBF + (long)t * 64, (long)Tc * 64, P_WihT, nullptr, 1.f, G1, nullptr, 512, Nc, 64, 512, 0);
        gemm(hbf, 128, P_WhhT, nullptr, 1.f, G2, nullptr, 512, Nc, 128, 512, 0);
        k_lstm_cell<<<gdiv(CH,256),256,0,stream>>>(G1, G2, enc_bih, enc_bhh, hcur, ccur, hbf, hsum, Nc);
    }
    k_scale<<<gdiv(CH,256),256,0,stream>>>(hsum, 1.0f/Tc, ht, CH);

    // -------- 6) temporal embedding + tr LSTM --------
    k_emb_gather<<<gdiv((long)Bc*Tc*70,256),256,0,stream>>>(weeks, minutes, week_tab, minute_tab, emb);
    k_small_lstm<<<Bc,512,0,stream>>>(emb, 70, tr_Wih, tr_Whh, tr_bih, tr_bhh, lth);
    k_mean_T<<<gdiv((long)Bc*Hc,256),256,0,stream>>>(lth, gth, Bc, Hc);

    // -------- 7) region graph conv --------
    zero(rdego, Rc); zero(rdegi, Rc);
    k_deg<<<gdiv(REc,256),256,0,stream>>>(rsrc, rdst, rdego, rdegi, REc);
    k_scalecast<<<gdiv((long)Rc*Hc,256),256,0,stream>>>(region_tab, rdego, rxs, Hc, (long)Rc*Hc);
    gemm(rxs, 128, P_ggW, nullptr, 1.f, rh, nullptr, 128, Rc, 128, 128, 0);
    zero(ragg, (long)Rc * Hc);
    k_scatter<<<REc,256,0,stream>>>(rsrc, rdst, rh, ragg, Hc);
    k_conv_final<<<gdiv((long)Rc*Hc,256),256,0,stream>>>(ragg, rdegi, gg_b, gemb, 128, 128, (long)Rc*Hc, 1);
    k_gather_rows<<<gdiv((long)Bc*Hc,256),256,0,stream>>>(gemb, gsi, gsh, Hc, Bc);

    // -------- 8) c_ht = et-MLP(lth) + es-MLP(gsh) --------
    k_naive_gemm<<<gdiv((long)Bc*Tc*256,256),256,0,stream>>>(lth, 128, et_W1, 0, et_b1, eth, 256, Bc*Tc, 128, 256, 1);
    k_naive_gemm<<<gdiv((long)Bc*Tc*64,256),256,0,stream>>>(eth, 256, et_W2, 0, et_b2, eto, 64, Bc*Tc, 256, 64, 0);
    k_naive_gemm<<<gdiv((long)Bc*256,256),256,0,stream>>>(gsh, 128, es_W1, 0, es_b1, esh, 256, Bc, 128, 256, 1);
    k_naive_gemm<<<gdiv((long)Bc*768,256),256,0,stream>>>(esh, 256, es_W2, 0, es_b2, eso, 768, Bc, 256, 768, 0);
    k_cht_add<<<gdiv((long)Bc*Tc*TCDc,256),256,0,stream>>>(eto, eso, cht);

    // -------- 9) lsh graph conv; c_hs --------
    k_scalecast_lsh<<<gdiv((long)Nc*160,256),256,0,stream>>>(lsf, link_tab, lsi, deg_out, xb_lsh);
    gemm(xb_lsh, 160, P_lgW, nullptr, 1.f, BIGh, nullptr, 128, Nc, 160, 128, 0);
    zero(BIGa, CH);
    k_scatter<<<Ec,256,0,stream>>>(src, dst, BIGh, BIGa, 128);
    k_conv_final<<<gdiv(CH,256),256,0,stream>>>(BIGa, deg_in, lg_b, lshb, 128, 128, CH, 1);
    k_chs<<<gdiv(CH,256),256,0,stream>>>(gth, lbi, lshb, c_hs);

    // -------- 10) decoder graph convs on ht_d = [ht, c_hs] --------
    k_scalecast_htd<<<gdiv((long)Nc*256,256),256,0,stream>>>(ht, c_hs, deg_out, xb_htd);
    gemm(xb_htd, 256, P_decW0, nullptr, 1.f, BIGh, nullptr, 128, Nc, 256, 128, 0);
    zero(BIGa, CH);
    k_scatter<<<Ec,256,0,stream>>>(src, dst, BIGh, BIGa, 128);
    k_conv_final<<<gdiv(CH,256),256,0,stream>>>(BIGa, deg_in, dec_b0, htd1, 128, 128, CH, 1);
    k_scalecast<<<gdiv(CH,256),256,0,stream>>>(htd1, deg_out, xb_dc, 128, CH);
    gemm(xb_dc, 128, P_decW1, nullptr, 1.f, BIGh, nullptr, 128, Nc, 128, 128, 0);
    zero(BIGa, CH);
    k_scatter<<<Ec,256,0,stream>>>(src, dst, BIGh, BIGa, 128);
    k_conv_final<<<gdiv(CH,256),256,0,stream>>>(BIGa, deg_in, dec_b1, htd2, 128, 128, CH, 1);

    // -------- 11) dec LSTM; alpha, beta --------
    k_concat_hsd<<<gdiv((long)Bc*Tc*192,256),256,0,stream>>>(hsB, cht, hsd);
    k_small_lstm<<<Bc,512,0,stream>>>(hsd, 192, dec_Wih, dec_Whh, dec_bih, dec_bhh, decH);
    k_add<<<gdiv((long)Bc*Tc*Hc,256),256,0,stream>>>(decH, hsB, alpha, (long)Bc*Tc*Hc);
    k_beta<<<gdiv(CH,256),256,0,stream>>>(htd2, ht, beta, xb_beta);

    // -------- 12) attention (algebraically collapsed) --------
    // t2s side: ka, va from beta; u = ka @ Wq^T; vo = va @ Wo
    gemm(xb_beta, 128, P_t2sWk, t2s_bk, 1.f, ka, ka_bf, 256, Nc, 128, 256, 0);
    gemm(xb_beta, 128, P_t2sWv, t2s_bv, 1.f, va, va_bf, 256, Nc, 128, 256, 0);
    gemm(ka_bf, 256, P_t2sWqT, nullptr, 1.f, ubuf, nullptr, 128, Nc, 256, 128, 0);
    gemm(va_bf, 256, P_t2sWo, nullptr, 1.f, vo, nullptr, 256, Nc, 256, 256, 0);
    // s2t side: softmax over size-1 axis == 1 -> beta_nn = (sum_t alpha_new)@Wv + T*bv, then @Wo+bo
    k_asum_bf<<<gdiv(CH,256),256,0,stream>>>(alpha, lbi, asum_bf);
    gemm(asum_bf, 128, P_s2tWv, s2t_bv, (float)Tc, vbs, vbs_bf, 256, Nc, 128, 256, 0);
    gemm(vbs_bf, 256, P_s2tWo, s2t_bo, 1.f, nullptr, bnn_bf, 256, Nc, 256, 256, 0);
    // eb MLP: beta_ext written directly into d_out (N,3072) == (N,T,256)
    gemm(bnn_bf, 256, P_ebW1, eb_b1, 1.f, nullptr, hb_bf, 256, Nc, 256, 256, 1);
    gemm(hb_bf, 256, P_ebW2, eb_b2, 1.f, out, nullptr, 3072, Nc, 256, 3072, 0);
    // scores + softmax over T, then out += w * vo + bo
    k_scores<<<Nc,32,0,stream>>>(alpha, lbi, ubuf, ka, t2s_bq, wbuf);
    k_final_out<<<Nc*Tc,256,0,stream>>>(out, wbuf, vo, t2s_bo);
}
